// SS2DChannel_31516470018307
// MI455X (gfx1250) — compile-verified, matrix operand order checked by reading, not run on hardware
//
#include <hip/hip_runtime.h>
#include <math.h>

// ---------------------------------------------------------------------------
// SS2D channel-scan block for MI455X (gfx1250, wave32, WMMA).
//
// Precision: bf16 WMMA (v_wmma_f32_16x16x32_bf16) with fp32 accumulation for
// all GEMMs; fp32 VALU for softplus / exp-scan / layernorm / silu.
// ---------------------------------------------------------------------------

#define D_MODEL  768
#define D_INNER  1536
#define D_STATE  16
#define DT_RANK  48
#define BATCH    1024
#define KDIR     2

typedef __bf16 bf16_t;
typedef __bf16 bf16x8 __attribute__((ext_vector_type(8)));
typedef __bf16 v16bf  __attribute__((ext_vector_type(16)));
typedef float  v8f    __attribute__((ext_vector_type(8)));

// ---------------------------------------------------------------------------
// Generic bf16-WMMA GEMM:  C[M,N] = A[M,K] * B[N,K]^T   (A,B fp32 row-major,
// converted to bf16 while staging into LDS; fp32 accumulate; epilogue functor)
//
// Block tile 128x128, 8 waves (4x2), wave tile 32x64 (2x4 WMMA tiles),
// K chunk 32, double-buffered LDS: next chunk's global_load_b128 are issued
// into register arrays before the current chunk's WMMAs, so the loadcnt wait
// lands after a full chunk of compute.  global_prefetch_b8 warms chunk +2.
// ---------------------------------------------------------------------------
static constexpr int BM = 128;    // block tile M
static constexpr int BN = 128;    // block tile N
static constexpr int BK = 32;     // K chunk (one WMMA K step)
static constexpr int SA = BK + 8; // LDS row stride (bf16 elems) -> 80B rows

__device__ inline v16bf load_frag(const bf16_t* row, int hi) {
    // ISA 7.12.2 16-bit operand layout: lane(r,hi) elements 0..7 <-> K=hi*8+i,
    // elements 8..15 <-> K=16+hi*8+i
    bf16x8 c0 = *(const bf16x8*)(row + hi * 8);
    bf16x8 c1 = *(const bf16x8*)(row + 16 + hi * 8);
    v16bf f;
    #pragma unroll
    for (int i = 0; i < 8; ++i) { f[i] = c0[i]; f[8 + i] = c1[i]; }
    return f;
}

template <class Epi, bool ALIGNED>
__global__ __launch_bounds__(256)
void gemm_bf16_kernel(const float* __restrict__ A, int lda,
                      const float* __restrict__ B, int ldb,
                      int M, int N, int Kd, Epi epi)
{
    __shared__ bf16_t As[2][BM * SA];
    __shared__ bf16_t Bs[2][BN * SA];

    const int tid  = threadIdx.x;
    const int lane = tid & 31;
    const int r    = lane & 15;     // row within 16 (A: M-row, B: N-row, C: N-col)
    const int hi   = lane >> 4;     // half-wave selector
    const int w    = tid >> 5;      // wave id 0..7
    const int wm   = (w >> 1) * 32; // wave M offset: 0,32,64,96
    const int wn   = (w & 1) * 64;  // wave N offset: 0,64
    const int m0   = blockIdx.y * BM;
    const int n0   = blockIdx.x * BN;

    // fixed per-thread staging geometry: 4 float4 slots each for A and B
    const int srow = tid >> 3;        // base row for it=0 (rows advance by 32/it)
    const int scol = (tid & 7) * 4;   // K sub-column

    v8f acc[2][4] = {};

    // issue next chunk's global loads into register arrays (no wait here)
    auto load_chunk = [&](int kc, float4* va, float4* vb) {
        #pragma unroll
        for (int it = 0; it < 4; ++it) {
            const int row = srow + it * 32;
            const int gm  = m0 + row;
            const int gk  = kc + scol;
            va[it] = make_float4(0.f, 0.f, 0.f, 0.f);
            if (ALIGNED || (gm < M && gk < Kd))
                va[it] = *(const float4*)(A + (size_t)gm * lda + gk);
        }
        #pragma unroll
        for (int it = 0; it < 4; ++it) {
            const int row = srow + it * 32;
            const int gn  = n0 + row;
            const int gk  = kc + scol;
            vb[it] = make_float4(0.f, 0.f, 0.f, 0.f);
            if (ALIGNED || (gn < N && gk < Kd))
                vb[it] = *(const float4*)(B + (size_t)gn * ldb + gk);
        }
    };

    // fp32 -> bf16 convert + LDS store
    auto store_chunk = [&](int buf, const float4* va, const float4* vb) {
        #pragma unroll
        for (int it = 0; it < 4; ++it) {
            bf16_t* d = &As[buf][(srow + it * 32) * SA + scol];
            d[0] = (bf16_t)va[it].x; d[1] = (bf16_t)va[it].y;
            d[2] = (bf16_t)va[it].z; d[3] = (bf16_t)va[it].w;
        }
        #pragma unroll
        for (int it = 0; it < 4; ++it) {
            bf16_t* d = &Bs[buf][(srow + it * 32) * SA + scol];
            d[0] = (bf16_t)vb[it].x; d[1] = (bf16_t)vb[it].y;
            d[2] = (bf16_t)vb[it].z; d[3] = (bf16_t)vb[it].w;
        }
    };

    // fragments + 8 WMMA on one LDS buffer
    auto compute = [&](int buf) {
        v16bf af[2], bfrag[4];
        #pragma unroll
        for (int mt = 0; mt < 2; ++mt)
            af[mt] = load_frag(&As[buf][(wm + mt * 16 + r) * SA], hi);
        #pragma unroll
        for (int nt = 0; nt < 4; ++nt)
            bfrag[nt] = load_frag(&Bs[buf][(wn + nt * 16 + r) * SA], hi);
        #pragma unroll
        for (int mt = 0; mt < 2; ++mt)
            #pragma unroll
            for (int nt = 0; nt < 4; ++nt)
                acc[mt][nt] = __builtin_amdgcn_wmma_f32_16x16x32_bf16(
                    false, af[mt], false, bfrag[nt],
                    (short)0, acc[mt][nt], false, false);
    };

    // ---- prologue: stage chunk 0 ----
    float4 va[4], vb[4];
    load_chunk(0, va, vb);
    store_chunk(0, va, vb);
    __syncthreads();

    int cur = 0;
    for (int kc = 0; kc < Kd; kc += BK) {
        const bool more = (kc + BK) < Kd;
        if (more) load_chunk(kc + BK, va, vb);    // in-flight during compute

        // prefetch chunk +2 into cache (global_prefetch_b8)
        if (kc + 2 * BK < Kd) {
            const int row_a = tid >> 1;           // 2 threads cover a 32-elem row
            const int cpa   = (tid & 1) * 16;
            if (ALIGNED || m0 + row_a < M)
                __builtin_prefetch(A + (size_t)(m0 + row_a) * lda + kc + 2 * BK + cpa, 0, 3);
            if (ALIGNED || n0 + row_a < N)
                __builtin_prefetch(B + (size_t)(n0 + row_a) * ldb + kc + 2 * BK + cpa, 0, 3);
        }

        compute(cur);                             // ds_load + wmma on cur buffer

        if (more) {
            __syncthreads();                      // everyone done reading cur^1
            store_chunk(cur ^ 1, va, vb);         // loadcnt wait lands here
            __syncthreads();                      // stores visible
        }
        cur ^= 1;
    }

    // ---- epilogue: C layout -> lane(r,hi), vgpr g holds row (g + 8*hi), col r
    #pragma unroll
    for (int mt = 0; mt < 2; ++mt) {
        #pragma unroll
        for (int nt = 0; nt < 4; ++nt) {
            const int cn = n0 + wn + nt * 16 + r;
            #pragma unroll
            for (int g = 0; g < 8; ++g) {
                const int cm = m0 + wm + mt * 16 + g + 8 * hi;
                if (ALIGNED || (cm < M && cn < N)) epi.store(cm, cn, acc[mt][nt][g]);
            }
        }
    }
}

// ---- epilogues -------------------------------------------------------------
struct EpiXZ {               // split xz: cols [0,1536) -> xs ; [1536,2304) -> silu
    float* xs;
    float* sil;
    __device__ void store(int m, int n, float v) const {
        if (n < D_INNER) {
            xs[(size_t)m * D_INNER + n] = v;
        } else {
            sil[(size_t)m * D_MODEL + (n - D_INNER)] = v / (1.f + expf(-v));
        }
    }
};
struct EpiPlain {            // plain fp32 store
    float* c; int ldc;
    __device__ void store(int m, int n, float v) const {
        c[(size_t)m * ldc + n] = v;
    }
};
struct EpiSoftplus {         // delta = softplus(v + bias[n])
    float* c; int ldc; const float* bias;
    __device__ void store(int m, int n, float v) const {
        float t = v + bias[n];
        c[(size_t)m * ldc + n] = (t > 20.f) ? t : log1pf(expf(t));
    }
};
struct EpiMulSil {           // out = v * sil[m,n]
    float* c; const float* sil;
    __device__ void store(int m, int n, float v) const {
        c[(size_t)m * D_MODEL + n] = v * sil[(size_t)m * D_MODEL + n];
    }
};

// ---------------------------------------------------------------------------
// A = -exp(A_logs)
// ---------------------------------------------------------------------------
__global__ void k_negexp(const float* __restrict__ in, float* __restrict__ out,
                         int n) {
    int i = blockIdx.x * blockDim.x + threadIdx.x;
    if (i < n) out[i] = -expf(in[i]);
}

// ---------------------------------------------------------------------------
// Channel scan: 16 lanes per (b,k) pair, one lane per state n.
//   h_n <- exp(delta_d * A[k,d,n]) * h_n + (delta_d * u_d) * B_n
//   y[b,k,d] = sum_n h_n * C_n + Ds[k,d] * u_d
// A[k] staged through LDS in d-chunks (broadcast across the 16 b-rows/block).
// ---------------------------------------------------------------------------
static constexpr int SCAN_CHUNK = 128;

__global__ __launch_bounds__(256)
void k_scan(const float* __restrict__ delta,  // [B, 2*D_INNER]
            const float* __restrict__ xs,     // [B, D_INNER]  (u)
            const float* __restrict__ xdbl,   // [B, 160]  (B at +48, C at +64 per k*80)
            const float* __restrict__ Aneg,   // [2, D_INNER, 16]
            const float* __restrict__ Ds,     // [2*D_INNER]
            float* __restrict__ y)            // [B, 2, D_INNER]
{
    const int k   = blockIdx.y;
    const int tid = threadIdx.x;
    const int n   = tid & 15;
    const int b   = blockIdx.x * 16 + (tid >> 4);

    __shared__ float Ach[SCAN_CHUNK * D_STATE];

    const float Bn = xdbl[(size_t)b * 160 + k * 80 + DT_RANK + n];
    const float Cn = xdbl[(size_t)b * 160 + k * 80 + DT_RANK + D_STATE + n];
    const float* drow = delta + (size_t)b * (2 * D_INNER) + k * D_INNER;
    const float* urow = xs + (size_t)b * D_INNER;
    const float* Dsk  = Ds + k * D_INNER;
    float* yrow = y + (size_t)b * (2 * D_INNER) + k * D_INNER;

    float h = 0.f;
    for (int d0 = 0; d0 < D_INNER; d0 += SCAN_CHUNK) {
        __syncthreads();
        for (int i = tid; i < SCAN_CHUNK * D_STATE; i += 256)
            Ach[i] = Aneg[((size_t)k * D_INNER + d0) * D_STATE + i];
        __syncthreads();
        for (int dd = 0; dd < SCAN_CHUNK; ++dd) {
            const int d  = d0 + dd;
            const float dl = drow[d];
            const float u  = urow[d];
            const float a  = Ach[dd * D_STATE + n];
            const float e  = expf(dl * a);
            h = e * h + (dl * u) * Bn;
            float v = h * Cn;
            v += __shfl_xor(v, 1, 16);
            v += __shfl_xor(v, 2, 16);
            v += __shfl_xor(v, 4, 16);
            v += __shfl_xor(v, 8, 16);
            if (n == 0) yrow[d] = v + Dsk[d] * u;
        }
    }
}

// ---------------------------------------------------------------------------
// y = LayerNorm(y[:,0,:] + y[:,1,:]) * w + b   (one block per batch row)
// ---------------------------------------------------------------------------
__global__ __launch_bounds__(256)
void k_norm(const float* __restrict__ y,      // [B, 2, D_INNER]
            const float* __restrict__ w,
            const float* __restrict__ bias,
            float* __restrict__ out)          // [B, D_INNER]
{
    const int b   = blockIdx.x;
    const int tid = threadIdx.x;
    __shared__ float r1[256];
    __shared__ float r2[256];

    const float* y0 = y + (size_t)b * (2 * D_INNER);
    const float* y1 = y0 + D_INNER;

    float s = 0.f, s2 = 0.f;
    for (int d = tid; d < D_INNER; d += 256) {
        float v = y0[d] + y1[d];
        s += v; s2 += v * v;
    }
    r1[tid] = s; r2[tid] = s2;
    __syncthreads();
    for (int off = 128; off > 0; off >>= 1) {
        if (tid < off) { r1[tid] += r1[tid + off]; r2[tid] += r2[tid + off]; }
        __syncthreads();
    }
    const float mu  = r1[0] * (1.f / D_INNER);
    const float var = r2[0] * (1.f / D_INNER) - mu * mu;
    const float inv = rsqrtf(var + 1e-5f);
    for (int d = tid; d < D_INNER; d += 256) {
        float v = y0[d] + y1[d];
        out[(size_t)b * D_INNER + d] = (v - mu) * inv * w[d] + bias[d];
    }
}

// ---------------------------------------------------------------------------
// Launch pipeline
// ---------------------------------------------------------------------------
extern "C" void kernel_launch(void* const* d_in, const int* in_sizes, int n_in,
                              void* d_out, int out_size, void* d_ws, size_t ws_size,
                              hipStream_t stream) {
    const float* x      = (const float*)d_in[0]; // [1024, 768]
    const float* w_in   = (const float*)d_in[1]; // [3072, 768]
    const float* x_proj = (const float*)d_in[2]; // [2, 80, 1536]
    const float* dt_w   = (const float*)d_in[3]; // [2, 1536, 48]
    const float* dt_b   = (const float*)d_in[4]; // [2, 1536]
    const float* A_logs = (const float*)d_in[5]; // [3072, 16]
    const float* Ds     = (const float*)d_in[6]; // [3072]
    const float* nw     = (const float*)d_in[7]; // [1536]
    const float* nb     = (const float*)d_in[8]; // [1536]
    const float* w_out  = (const float*)d_in[9]; // [768, 1536]
    float* out = (float*)d_out;                  // [1024, 768]

    // workspace layout (fp32), ~41.7 MB total
    float* ws    = (float*)d_ws;
    float* xs    = ws;                               // 1024*1536
    float* sil   = xs    + (size_t)BATCH * D_INNER;  // 1024*768
    float* xdbl  = sil   + (size_t)BATCH * D_MODEL;  // 1024*160
    float* delta = xdbl  + (size_t)BATCH * 160;      // 1024*3072
    float* Aneg  = delta + (size_t)BATCH * 2 * D_INNER; // 3072*16
    float* ybuf  = Aneg  + (size_t)2 * D_INNER * D_STATE; // 1024*3072
    float* ynorm = ybuf  + (size_t)BATCH * 2 * D_INNER;   // 1024*1536

    // 0) A = -exp(A_logs)
    k_negexp<<<(2 * D_INNER * D_STATE + 255) / 256, 256, 0, stream>>>(
        A_logs, Aneg, 2 * D_INNER * D_STATE);

    // 1) xz GEMM, N truncated to 2304 (z cols 768..1535 are unused by the ref)
    //    M=1024, N=2304, K=768 -> all tile-aligned
    gemm_bf16_kernel<EpiXZ, true><<<dim3((D_INNER + D_MODEL) / BN, BATCH / BM), 256, 0, stream>>>(
        x, D_MODEL, w_in, D_MODEL, BATCH, D_INNER + D_MODEL, D_MODEL,
        EpiXZ{xs, sil});

    // 2) x_dbl = xs @ x_proj^T   [1024,160]  (N unaligned -> guarded)
    gemm_bf16_kernel<EpiPlain, false><<<dim3((160 + BN - 1) / BN, BATCH / BM), 256, 0, stream>>>(
        xs, D_INNER, x_proj, D_INNER, BATCH, 160, D_INNER,
        EpiPlain{xdbl, 160});

    // 3) delta = softplus(dt_low @ dt_w^T + dt_b)  per direction k (K=48 -> guarded)
    for (int k = 0; k < KDIR; ++k) {
        gemm_bf16_kernel<EpiSoftplus, false><<<dim3(D_INNER / BN, BATCH / BM), 256, 0, stream>>>(
            xdbl + k * 80, 160,
            dt_w + (size_t)k * D_INNER * DT_RANK, DT_RANK,
            BATCH, D_INNER, DT_RANK,
            EpiSoftplus{delta + k * D_INNER, 2 * D_INNER, dt_b + k * D_INNER});
    }

    // 4) channel scan
    k_scan<<<dim3(BATCH / 16, KDIR), 256, 0, stream>>>(
        delta, xs, xdbl, Aneg, Ds, ybuf);

    // 5) layernorm of y0+y1
    k_norm<<<BATCH, 256, 0, stream>>>(ybuf, nw, nb, ynorm);

    // 6) out = (ynorm @ w_out^T) * silu(z[:, :768])   M/N/K tile-aligned
    gemm_bf16_kernel<EpiMulSil, true><<<dim3(D_MODEL / BN, BATCH / BM), 256, 0, stream>>>(
        ynorm, D_INNER, w_out, D_INNER, BATCH, D_MODEL, D_INNER,
        EpiMulSil{out, sil});
}
